// ImprovedHeteroGAT_57724360458711
// MI455X (gfx1250) — matrix operand order repaired and checked by reading, not compiled, not run on hardware
//
#include <hip/hip_runtime.h>
#include <stdint.h>
#include <stddef.h>

// ---------------- problem constants (match reference) ----------------
#define NLIG   20000
#define NTGT   2000
#define NEDGE  40000
#define LIN_D  1024
#define TIN_D  1280
#define HC_D   2048      // H*C
#define HH     8
#define CC     256

typedef __bf16 bf16;
typedef __attribute__((ext_vector_type(16))) __bf16 v16bf;
typedef __attribute__((ext_vector_type(8)))  float  v8f;

struct __align__(16) U128 { unsigned u[4]; };
union FR { U128 q[2]; v16bf v; };   // one 32-byte WMMA operand (8 VGPRs)

__device__ __forceinline__ unsigned short f2bf_bits(float f) {
  unsigned u = __builtin_bit_cast(unsigned, f);
  unsigned r = u + 0x7fffu + ((u >> 16) & 1u);   // round-to-nearest-even
  return (unsigned short)(r >> 16);
}
__device__ __forceinline__ bf16 f2bf(float f) {
  unsigned short us = f2bf_bits(f);
  return __builtin_bit_cast(bf16, us);
}
__device__ __forceinline__ float bf2f(bf16 h) {
  unsigned short us = __builtin_bit_cast(unsigned short, h);
  unsigned u = ((unsigned)us) << 16;
  return __builtin_bit_cast(float, u);
}

__device__ __forceinline__ float wave_reduce_add(float s) {
  #pragma unroll
  for (int off = 16; off > 0; off >>= 1) s += __shfl_down(s, off);
  return s;
}

__device__ __forceinline__ void atomicMaxF(float* addr, float v) {
  if (v >= 0.f) atomicMax((int*)addr, __builtin_bit_cast(int, v));
  else          atomicMin((unsigned*)addr, __builtin_bit_cast(unsigned, v));
}

// ---- CDNA5 async global->LDS copy (ASYNCcnt path) ----
__device__ __forceinline__ void async_ld_b128(unsigned lds_off, const void* g) {
  unsigned long long ga = (unsigned long long)(uintptr_t)g;
  asm volatile("global_load_async_to_lds_b128 %0, %1, off" :: "v"(lds_off), "v"(ga) : "memory");
}
__device__ __forceinline__ void wait_async_le4() { asm volatile("s_wait_asynccnt 0x4" ::: "memory"); }
__device__ __forceinline__ void wait_async_0()  { asm volatile("s_wait_asynccnt 0x0" ::: "memory"); }

// ---------------- elementwise helpers ----------------
__global__ __launch_bounds__(256) void k_cvt_bf16(const float* __restrict__ in,
                                                  bf16* __restrict__ out, size_t n) {
  size_t i = (size_t)blockIdx.x * 256 + threadIdx.x;
  if (i < n) out[i] = f2bf(in[i]);
}

__global__ __launch_bounds__(256) void k_fill(float* __restrict__ p, float v, size_t n) {
  size_t i = (size_t)blockIdx.x * 256 + threadIdx.x;
  if (i < n) p[i] = v;
}

// ---------------- pack f32 weight [K,N] -> WMMA-fragment-ready bf16 ----------------
// Layout: chunk = (kb, sb) with kb = K/32 blocks, sb = N/16 subtiles.
// Within chunk: 32 lanes x 8 dwords (contiguous 32B per lane).
// Lane l covers n = sb*16 + (l&15), K-halves kbase = kb*32 + (l>>4)*16;
// dword v holds bf16 pair (K = kbase+2v, kbase+2v+1).
__global__ __launch_bounds__(256) void k_pack_b(const float* __restrict__ W,
                                                unsigned* __restrict__ out,
                                                int K, int N) {
  size_t gid = (size_t)blockIdx.x * 256 + threadIdx.x;
  size_t total = (size_t)K * N / 2;
  if (gid >= total) return;
  unsigned v    = gid & 7;
  unsigned lane = (gid >> 3) & 31;
  size_t  chunk = gid >> 8;
  int npc = N >> 4;
  int sb = (int)(chunk % npc);
  int kb = (int)(chunk / npc);
  int n  = sb * 16 + (lane & 15);
  int k0 = kb * 32 + (int)(lane >> 4) * 16 + 2 * v;
  unsigned b0 = f2bf_bits(W[(size_t)k0 * N + n]);
  unsigned b1 = f2bf_bits(W[(size_t)(k0 + 1) * N + n]);
  out[gid] = b0 | (b1 << 16);
}

// ---------------- bf16 WMMA GEMM: C[M,N] = A[M,K] @ B[K,N] ----------------
// A: bf16 row-major. Bp: pre-packed fragment layout (see k_pack_b). C: f32.
// Requires N%128==0, K%32==0; M bounds-handled.
// Block 256 threads = 8 waves (4 M-groups x 2 N-groups).
// Tile 128(M) x 128(N) x 32(K); per wave: 2 A-frags x 4 B-frags = 8 WMMA / K-step.
// A+B tiles double-buffered in LDS via global_load_async_to_lds_b128.
#define BM 128
#define BN 128
#define BK 32
__global__ __launch_bounds__(256) void k_gemm_bf16(const bf16* __restrict__ A,
                                                   const unsigned* __restrict__ Bp,
                                                   float* __restrict__ C,
                                                   int M, int N, int K) {
  __shared__ __align__(16) bf16     As[2][BM][BK];   // 2 x 8 KB
  __shared__ __align__(16) unsigned Bs[2][2048];     // 2 x 8 KB (8 packed chunks)

  const int tid  = threadIdx.x;
  const int wid  = tid >> 5;
  const int lane = tid & 31;
  const int bm   = blockIdx.y * BM;
  const int bn   = blockIdx.x * BN;
  const int wm   = (wid & 3) * 32;   // wave M offset (2 frags of 16)
  const int wn   = (wid >> 2) * 64;  // wave N offset (4 subtiles of 16)

  v8f acc[2][4] = {};

  // ---- async stage helpers ----
  // A tile: 128 rows x 64B; 256 threads x 32B each (2 x b128).
  const int a_row  = tid >> 1;
  const int a_colb = (tid & 1) * 32;     // byte offset within 64B row
  int a_grow = bm + a_row; if (a_grow >= M) a_grow = M - 1;  // clamp (rows >= M never stored)
  const char* a_src_base = (const char*)(A + (size_t)a_grow * K) + a_colb;
  // B tile: 8 contiguous packed chunks = 8 KB; 256 threads x 32B each.
  const size_t b_chunk_row = (size_t)(bn >> 4) * 256;  // dword offset of first chunk at kb

  auto issue_tile = [&](int buf, int kk) {
    {
      unsigned l0 = (unsigned)(uintptr_t)((char*)&As[buf][a_row][0] + a_colb);
      const char* g = a_src_base + (size_t)kk * 2;  // advance kk bf16 elems
      async_ld_b128(l0, g);
      async_ld_b128(l0 + 16, g + 16);
    }
    {
      const char* g = (const char*)(Bp + (size_t)(kk >> 5) * (N >> 4) * 256 + b_chunk_row) + tid * 16;
      unsigned l0 = (unsigned)(uintptr_t)((char*)&Bs[buf][0] + tid * 16);
      async_ld_b128(l0, g);
      async_ld_b128(l0 + 4096, g + 4096);
    }
  };

  const int nt = K / BK;
  int cur = 0;
  issue_tile(0, 0);

  for (int i = 0; i < nt; ++i) {
    const bool has_next = (i + 1 < nt);
    if (has_next) issue_tile(cur ^ 1, (i + 1) * BK);
    if (has_next) wait_async_le4(); else wait_async_0();
    __syncthreads();

    // ---- A fragments: rows wm + a*16 + (lane&15); K-half select by lane group ----
    FR af[2];
    const int kb_b = (lane >> 4) * 32;  // byte offset of K-half within 64B row
    #pragma unroll
    for (int a = 0; a < 2; ++a) {
      const char* arow = (const char*)&As[cur][wm + a * 16 + (lane & 15)][0];
      af[a].q[0] = *(const U128*)(arow + kb_b);
      af[a].q[1] = *(const U128*)(arow + 16 + kb_b);
    }
    // ---- B fragments: packed chunk j = wn/16 + s, lane-contiguous 32B ----
    FR bf4[4];
    #pragma unroll
    for (int s = 0; s < 4; ++s) {
      const char* bptr = (const char*)&Bs[cur][0] + ((wn >> 4) + s) * 1024 + lane * 32;
      bf4[s].q[0] = *(const U128*)(bptr);
      bf4[s].q[1] = *(const U128*)(bptr + 16);
    }
    // ---- 8 independent WMMAs ----
    #pragma unroll
    for (int a = 0; a < 2; ++a)
      #pragma unroll
      for (int s = 0; s < 4; ++s)
        acc[a][s] = __builtin_amdgcn_wmma_f32_16x16x32_bf16(
            false, af[a].v, false, bf4[s].v, (short)0, acc[a][s], false, false);

    __syncthreads();   // all waves done reading buf before it is overwritten
    cur ^= 1;
  }

  // ---- store C: lane layout M = r + 8*(lane>=16), N = lane%16 ----
  #pragma unroll
  for (int a = 0; a < 2; ++a) {
    #pragma unroll
    for (int s = 0; s < 4; ++s) {
      const int col = bn + wn + s * 16 + (lane & 15);
      #pragma unroll
      for (int r8 = 0; r8 < 8; ++r8) {
        const int row = bm + wm + a * 16 + r8 + 8 * (lane >> 4);
        if (row < M) C[(size_t)row * N + col] = acc[a][s][r8];
      }
    }
  }
}

// ---------------- per-head dot: out[n,h] = sum_c X[n, h*256+c] * a[h*256+c] ----------------
__global__ __launch_bounds__(256) void k_headdot(const float* __restrict__ X,
                                                 const float* __restrict__ a,
                                                 float* __restrict__ out) {
  const int n = blockIdx.x;
  const int h = threadIdx.x >> 5;
  const int lane = threadIdx.x & 31;
  const float* xr = X + (size_t)n * HC_D + h * CC;
  const float* ar = a + h * CC;
  float s = 0.f;
  #pragma unroll
  for (int j = 0; j < 8; ++j) s += xr[lane + j * 32] * ar[lane + j * 32];
  s = wave_reduce_add(s);
  if (lane == 0) out[n * HH + h] = s;
}

// ---------------- matvec with per-head weights: out[n,h] = sum_i X[n,i] * wt[i,h] ----------------
template <typename T>
__global__ __launch_bounds__(256) void k_matvecH(const T* __restrict__ X, int K,
                                                 const float* __restrict__ wt,
                                                 float* __restrict__ out) {
  const int n = blockIdx.x;
  const int h = threadIdx.x >> 5;
  const int lane = threadIdx.x & 31;
  float s = 0.f;
  for (int i = lane; i < K; i += 32) {
    float xv;
    if constexpr (sizeof(T) == 2) xv = bf2f(__builtin_bit_cast(bf16, *(const unsigned short*)&X[(size_t)n * K + i]));
    else                          xv = ((const float*)X)[(size_t)n * K + i];
    s += xv * wt[i * HH + h];
  }
  s = wave_reduce_add(s);
  if (lane == 0) out[n * HH + h] = s;
}

// ---------------- edge softmax passes ----------------
__global__ __launch_bounds__(256) void k_edge_max(const int* __restrict__ src,
                                                  const int* __restrict__ dst,
                                                  const float* __restrict__ als,
                                                  const float* __restrict__ ald,
                                                  float* __restrict__ escore,
                                                  float* __restrict__ m) {
  int gid = blockIdx.x * 256 + threadIdx.x;
  if (gid >= NEDGE * HH) return;
  int e = gid >> 3, h = gid & 7;
  int s = src[e], d = dst[e];
  float v = als[s * HH + h] + ald[d * HH + h];
  v = (v > 0.f) ? v : 0.2f * v;  // leaky_relu slope 0.2
  escore[gid] = v;
  atomicMaxF(&m[d * HH + h], v);
}

__global__ __launch_bounds__(256) void k_edge_exp(const int* __restrict__ dst,
                                                  const float* __restrict__ escore,
                                                  const float* __restrict__ m,
                                                  float* __restrict__ den,
                                                  float* __restrict__ exb) {
  int gid = blockIdx.x * 256 + threadIdx.x;
  if (gid >= NEDGE * HH) return;
  int e = gid >> 3, h = gid & 7;
  int d = dst[e];
  float ex = __expf(escore[gid] - m[d * HH + h]);
  exb[gid] = ex;
  atomicAdd(&den[d * HH + h], ex);
}

// ---------------- weighted scatter: agg[dst] += hs[src] * alpha ----------------
__global__ __launch_bounds__(256) void k_scatter(const int* __restrict__ src,
                                                 const int* __restrict__ dst,
                                                 const float* __restrict__ hs,
                                                 const float* __restrict__ exb,
                                                 const float* __restrict__ den,
                                                 float* __restrict__ agg) {
  const int e = blockIdx.x;
  __shared__ float alpha[HH];
  const int s = src[e], d = dst[e];
  if (threadIdx.x < HH) alpha[threadIdx.x] = exb[e * HH + threadIdx.x] / den[d * HH + threadIdx.x];
  __syncthreads();
  #pragma unroll
  for (int j = 0; j < 8; ++j) {
    int idx = threadIdx.x + j * 256;
    atomicAdd(&agg[(size_t)d * HC_D + idx], hs[(size_t)s * HC_D + idx] * alpha[idx >> 8]);
  }
}

// ---------------- bias + ReLU + LayerNorm -> bf16 ----------------
__global__ __launch_bounds__(256) void k_relu_ln(const float* __restrict__ agg,
                                                 const float* __restrict__ bias,
                                                 const float* __restrict__ g,
                                                 const float* __restrict__ be,
                                                 bf16* __restrict__ out) {
  __shared__ float red[256];
  const int n = blockIdx.x, tid = threadIdx.x;
  float t[8];
  float s = 0.f;
  #pragma unroll
  for (int j = 0; j < 8; ++j) {
    int idx = tid + j * 256;
    float v = agg[(size_t)n * HC_D + idx] + bias[idx];
    v = v > 0.f ? v : 0.f;
    t[j] = v;
    s += v;
  }
  red[tid] = s; __syncthreads();
  for (int st = 128; st > 0; st >>= 1) { if (tid < st) red[tid] += red[tid + st]; __syncthreads(); }
  float mu = red[0] * (1.f / HC_D);
  __syncthreads();
  float vs = 0.f;
  #pragma unroll
  for (int j = 0; j < 8; ++j) { float d = t[j] - mu; vs += d * d; }
  red[tid] = vs; __syncthreads();
  for (int st = 128; st > 0; st >>= 1) { if (tid < st) red[tid] += red[tid + st]; __syncthreads(); }
  float rstd = rsqrtf(red[0] * (1.f / HC_D) + 1e-5f);
  #pragma unroll
  for (int j = 0; j < 8; ++j) {
    int idx = tid + j * 256;
    out[(size_t)n * HC_D + idx] = f2bf((t[j] - mu) * rstd * g[idx] + be[idx]);
  }
}

// ---------------- gather concat(l2[src], t2[dst]) -> ef bf16 [E,4096] ----------------
__global__ __launch_bounds__(256) void k_gather_ef(const bf16* __restrict__ lb,
                                                   const bf16* __restrict__ tb,
                                                   const int* __restrict__ src,
                                                   const int* __restrict__ dst,
                                                   bf16* __restrict__ ef) {
  const int e = blockIdx.x;
  const U128* lr = (const U128*)(lb + (size_t)src[e] * HC_D);
  const U128* tr = (const U128*)(tb + (size_t)dst[e] * HC_D);
  U128* o = (U128*)(ef + (size_t)e * (2 * HC_D));
  o[threadIdx.x]       = lr[threadIdx.x];
  o[256 + threadIdx.x] = tr[threadIdx.x];
}

// ---------------- final: out[e] = relu(h[e]+bm1) . Wm2 + bm2 ----------------
__global__ __launch_bounds__(256) void k_mlp2(const float* __restrict__ h,
                                              const float* __restrict__ bm1,
                                              const float* __restrict__ Wm2,
                                              const float* __restrict__ bm2,
                                              float* __restrict__ out, int E) {
  const int e = blockIdx.x * 8 + (threadIdx.x >> 5);
  const int lane = threadIdx.x & 31;
  if (e >= E) return;
  float s = 0.f;
  #pragma unroll
  for (int j = 0; j < 8; ++j) {
    int c = lane + j * 32;
    float v = h[(size_t)e * CC + c] + bm1[c];
    v = v > 0.f ? v : 0.f;
    s += v * Wm2[c];
  }
  s = wave_reduce_add(s);
  if (lane == 0) out[e] = s + bm2[0];
}

// ============================ host orchestration ============================
static inline size_t alignup(size_t x) { return (x + 255) & ~(size_t)255; }
static inline int cdiv(int a, int b) { return (a + b - 1) / b; }

extern "C" void kernel_launch(void* const* d_in, const int* in_sizes, int n_in,
                              void* d_out, int out_size, void* d_ws, size_t ws_size,
                              hipStream_t stream) {
  (void)in_sizes; (void)n_in; (void)out_size; (void)ws_size;
  const float* x_ligand = (const float*)d_in[0];
  const float* x_target = (const float*)d_in[1];
  const float* Ws1lt = (const float*)d_in[2];
  const float* Wd1lt = (const float*)d_in[3];
  const float* as1lt = (const float*)d_in[4];
  const float* ad1lt = (const float*)d_in[5];
  const float* b1lt  = (const float*)d_in[6];
  const float* Ws1tl = (const float*)d_in[7];
  const float* Wd1tl = (const float*)d_in[8];
  const float* as1tl = (const float*)d_in[9];
  const float* ad1tl = (const float*)d_in[10];
  const float* b1tl  = (const float*)d_in[11];
  const float* Ws2lt = (const float*)d_in[12];
  const float* Wd2lt = (const float*)d_in[13];
  const float* as2lt = (const float*)d_in[14];
  const float* ad2lt = (const float*)d_in[15];
  const float* b2lt  = (const float*)d_in[16];
  const float* Ws2tl = (const float*)d_in[17];
  const float* Wd2tl = (const float*)d_in[18];
  const float* as2tl = (const float*)d_in[19];
  const float* ad2tl = (const float*)d_in[20];
  const float* b2tl  = (const float*)d_in[21];
  const float* g1    = (const float*)d_in[22];
  const float* be1   = (const float*)d_in[23];
  const float* g2    = (const float*)d_in[24];
  const float* be2   = (const float*)d_in[25];
  const float* Wm1   = (const float*)d_in[26];
  const float* bm1   = (const float*)d_in[27];
  const float* Wm2   = (const float*)d_in[28];
  const float* bm2   = (const float*)d_in[29];
  const int* esrc    = (const int*)d_in[30];
  const int* edst    = (const int*)d_in[31];

  // ---- workspace carve-out ----
  char* w = (char*)d_ws;
  size_t off = 0;
  auto alloc = [&](size_t bytes) -> void* { void* p = w + off; off = alignup(off + bytes); return p; };

  bf16* xlb        = (bf16*)alloc((size_t)NLIG * LIN_D * 2);   // aliased by hmlp later
  bf16* xtb        = (bf16*)alloc((size_t)NTGT * TIN_D * 2);
  unsigned* Ws1ltp = (unsigned*)alloc((size_t)LIN_D * HC_D * 2);
  unsigned* Ws1tlp = (unsigned*)alloc((size_t)TIN_D * HC_D * 2);
  unsigned* Ws2ltp = (unsigned*)alloc((size_t)HC_D * HC_D * 2);
  unsigned* Ws2tlp = (unsigned*)alloc((size_t)HC_D * HC_D * 2);
  unsigned* Wm1p   = (unsigned*)alloc((size_t)2 * HC_D * CC * 2);
  float* hs_l   = (float*)alloc((size_t)NLIG * HC_D * 4);
  float* agg_l  = (float*)alloc((size_t)NLIG * HC_D * 4);   // contiguous after hs_l
  float* hs_t   = (float*)alloc((size_t)NTGT * HC_D * 4);
  float* agg_t  = (float*)alloc((size_t)NTGT * HC_D * 4);
  bf16* l1b     = (bf16*)alloc((size_t)NLIG * HC_D * 2);
  bf16* t1b     = (bf16*)alloc((size_t)NTGT * HC_D * 2);
  bf16* l2b     = (bf16*)alloc((size_t)NLIG * HC_D * 2);
  bf16* t2b     = (bf16*)alloc((size_t)NTGT * HC_D * 2);
  float* wtilde = (float*)alloc((size_t)HC_D * HH * 4);
  float* als    = (float*)alloc((size_t)NLIG * HH * 4);
  float* ald    = (float*)alloc((size_t)NLIG * HH * 4);
  float* mbuf   = (float*)alloc((size_t)NLIG * HH * 4);
  float* den    = (float*)alloc((size_t)NLIG * HH * 4);
  float* escore = (float*)alloc((size_t)NEDGE * HH * 4);
  float* exb    = (float*)alloc((size_t)NEDGE * HH * 4);
  bf16*  ef   = (bf16*)hs_l;   // E*4096*2 == |hs_l| + |agg_l|
  float* hmlp = (float*)xlb;   // E*256*4  == |xlb|

  auto cvt = [&](const float* in, bf16* out, size_t n) {
    k_cvt_bf16<<<dim3((unsigned)((n + 255) / 256)), dim3(256), 0, stream>>>(in, out, n);
  };
  auto pack = [&](const float* Wsrc, unsigned* out, int K, int N) {
    size_t total = (size_t)K * N / 2;
    k_pack_b<<<dim3((unsigned)((total + 255) / 256)), dim3(256), 0, stream>>>(Wsrc, out, K, N);
  };
  auto gemm = [&](const bf16* A, const unsigned* Bp, float* C, int M, int N, int K) {
    k_gemm_bf16<<<dim3(N / BN, cdiv(M, BM)), dim3(256), 0, stream>>>(A, Bp, C, M, N, K);
  };
  auto fill = [&](float* p, float v, size_t n) {
    k_fill<<<dim3((unsigned)((n + 255) / 256)), dim3(256), 0, stream>>>(p, v, n);
  };

  auto conv = [&](const bf16* srcb, int Ns, int Ksrc, const unsigned* Wsp,
                  const void* xdst, int Nd, int Kdst, bool dstBf16,
                  const float* Wd, const float* a_s, const float* a_d,
                  const int* sidx, const int* didx,
                  float* hs, float* agg) {
    gemm(srcb, Wsp, hs, Ns, HC_D, Ksrc);
    k_headdot<<<dim3(Ns), dim3(256), 0, stream>>>(hs, a_s, als);
    k_headdot<<<dim3(Kdst), dim3(256), 0, stream>>>(Wd, a_d, wtilde);
    if (dstBf16)
      k_matvecH<bf16><<<dim3(Nd), dim3(256), 0, stream>>>((const bf16*)xdst, Kdst, wtilde, ald);
    else
      k_matvecH<float><<<dim3(Nd), dim3(256), 0, stream>>>((const float*)xdst, Kdst, wtilde, ald);
    fill(mbuf, -3.0e38f, (size_t)Nd * HH);
    fill(den, 0.f, (size_t)Nd * HH);
    fill(agg, 0.f, (size_t)Nd * HC_D);
    const int eb = cdiv(NEDGE * HH, 256);
    k_edge_max<<<dim3(eb), dim3(256), 0, stream>>>(sidx, didx, als, ald, escore, mbuf);
    k_edge_exp<<<dim3(eb), dim3(256), 0, stream>>>(didx, escore, mbuf, den, exb);
    k_scatter<<<dim3(NEDGE), dim3(256), 0, stream>>>(sidx, didx, hs, exb, den, agg);
  };

  // ---- one-time precision conversion / weight packing ----
  cvt(x_ligand, xlb, (size_t)NLIG * LIN_D);
  cvt(x_target, xtb, (size_t)NTGT * TIN_D);
  pack(Ws1lt, Ws1ltp, LIN_D, HC_D);
  pack(Ws1tl, Ws1tlp, TIN_D, HC_D);
  pack(Ws2lt, Ws2ltp, HC_D, HC_D);
  pack(Ws2tl, Ws2tlp, HC_D, HC_D);
  pack(Wm1,   Wm1p,   2 * HC_D, CC);

  // ---- layer 1 ----
  conv(xlb, NLIG, LIN_D, Ws1ltp, x_target, NTGT, TIN_D, false,
       Wd1lt, as1lt, ad1lt, esrc, edst, hs_l, agg_t);
  conv(xtb, NTGT, TIN_D, Ws1tlp, x_ligand, NLIG, LIN_D, false,
       Wd1tl, as1tl, ad1tl, edst, esrc, hs_t, agg_l);
  k_relu_ln<<<dim3(NTGT), dim3(256), 0, stream>>>(agg_t, b1lt, g1, be1, t1b);
  k_relu_ln<<<dim3(NLIG), dim3(256), 0, stream>>>(agg_l, b1tl, g1, be1, l1b);

  // ---- layer 2 ----
  conv(l1b, NLIG, HC_D, Ws2ltp, t1b, NTGT, HC_D, true,
       Wd2lt, as2lt, ad2lt, esrc, edst, hs_l, agg_t);
  conv(t1b, NTGT, HC_D, Ws2tlp, l1b, NLIG, HC_D, true,
       Wd2tl, as2tl, ad2tl, edst, esrc, hs_t, agg_l);
  k_relu_ln<<<dim3(NTGT), dim3(256), 0, stream>>>(agg_t, b2lt, g2, be2, t2b);
  k_relu_ln<<<dim3(NLIG), dim3(256), 0, stream>>>(agg_l, b2tl, g2, be2, l2b);

  // ---- edge MLP ----
  k_gather_ef<<<dim3(NEDGE), dim3(256), 0, stream>>>(l2b, t2b, esrc, edst, ef);
  gemm(ef, Wm1p, hmlp, NEDGE, CC, 2 * HC_D);
  k_mlp2<<<dim3(cdiv(NEDGE, 8)), dim3(256), 0, stream>>>(hmlp, bm1, Wm2, bm2, (float*)d_out, NEDGE);
}
